// MyModel1_41205916238210
// MI455X (gfx1250) — matrix operand order repaired and checked by reading, not compiled, not run on hardware
//
#include <hip/hip_runtime.h>
#include <hip/hip_bf16.h>
#include <math.h>

typedef __attribute__((ext_vector_type(16))) _Float16 v16h;
typedef __attribute__((ext_vector_type(8)))  float    v8f;

constexpr int Bk = 512, Tk = 50, Pk = 64, Dk = 3;
constexpr int INT_H = 32, CONF_H = 16;
constexpr int ROWS  = Tk * (Pk - 1);      // 3150 interaction rows per batch
constexpr int NTILE = (ROWS + 15) / 16;   // 197 row-tiles of 16
constexpr int NWAVE = 8;                  // 256 threads = 8 wave32

static __device__ __forceinline__ v8f wmma16(v16h a, v16h b, v8f c) {
  // D(16x16,f32) = A(16x32,f16) x B(32x16,f16) + C  (bias rides in C)
  return __builtin_amdgcn_wmma_f32_16x16x32_f16(false, a, false, b, (short)0, c, false, false);
}

// lane <-> lane^16 exchange: DS_SWIZZLE group-of-32, xor=0x10, and=0x1F (SWAPX16)
static __device__ __forceinline__ float swap16(float x) {
  return __int_as_float(__builtin_amdgcn_ds_swizzle(__float_as_int(x), 0x401F));
}

// ACT: 0 = leaky(0.1) via max(x,0.1x), 1 = tanh, 2 = elu
template <int ACT>
static __device__ __forceinline__ float actf(float x) {
  if (ACT == 0) return fmaxf(x, 0.1f * x);
  if (ACT == 1) return tanhf(x);
  return x > 0.0f ? x : __expf(x) - 1.0f;
}

static __device__ __forceinline__ v16h zero16() {
  v16h z;
#pragma unroll
  for (int h = 0; h < 16; ++h) z[h] = (_Float16)0.0f;
  return z;
}

// W^T tile as a 16-bit WMMA A operand (16 out-features x 32 K).
// W is (Kmax x ldN) row-major f32; K(h,g) = h + 16g; K>=Kmax -> 0.
static __device__ __forceinline__ v16h loadA(const float* W, int ldN, int Fbase,
                                             int Kmax, int lane) {
  const int m = lane & 15;
  const int g = (lane >> 4) & 1;
  v16h a;
#pragma unroll
  for (int h = 0; h < 16; ++h) {
    const int k = h + 16 * g;
    const float w = (k < Kmax) ? W[k * ldN + Fbase + m] : 0.0f;
    a[h] = (_Float16)w;
  }
  return a;
}

// D-layout (two 16x16 f32 tiles = 32 features, bias pre-added via C)
// -> activation -> f16 B-layout for the next layer.
template <int ACT>
static __device__ __forceinline__ v16h repack32(v8f d0, v8f d1, int g) {
  float e0[8], e1[8], q0[8], q1[8];
#pragma unroll
  for (int r = 0; r < 8; ++r) { e0[r] = actf<ACT>(d0[r]); e1[r] = actf<ACT>(d1[r]); }
#pragma unroll
  for (int r = 0; r < 8; ++r) { q0[r] = swap16(e0[r]); q1[r] = swap16(e1[r]); }
  v16h out;
#pragma unroll
  for (int r = 0; r < 8; ++r) {
    out[r]     = (_Float16)(g ? q1[r] : e0[r]);
    out[r + 8] = (_Float16)(g ? e1[r] : q0[r]);
  }
  return out;
}

template <int ACT>
static __device__ __forceinline__ v16h repack16(v8f d0, int g) {
  float e0[8], q0[8];
#pragma unroll
  for (int r = 0; r < 8; ++r) e0[r] = actf<ACT>(d0[r]);
#pragma unroll
  for (int r = 0; r < 8; ++r) q0[r] = swap16(e0[r]);
  v16h out;
#pragma unroll
  for (int r = 0; r < 8; ++r) {
    out[r]     = (_Float16)(g ? 0.0f : e0[r]);
    out[r + 8] = (_Float16)(g ? 0.0f : q0[r]);
  }
  return out;
}

struct RowIn { v16h bv; float dirx, diry, wv; };

static __device__ __forceinline__ RowIn loadRow(const float* __restrict__ Xb,
                                                const float* __restrict__ wvt,
                                                int tile, int lane, int g) {
  RowIn ri;
  const int idx = tile * 16 + (lane & 15);
  const bool ld = (idx < ROWS) && (g == 0);   // only K=0..15 half carries data
  int t = 0, p = 1;
  if (idx < ROWS) { t = idx / (Pk - 1); p = 1 + idx % (Pk - 1); }
  float p0x = 0, p0y = 0, p0z = 0, pix = 0, piy = 0, piz = 0, wv = 0;
  if (ld) {
    const float* r = Xb + (size_t)t * (Pk * Dk);
    p0x = r[0];         p0y = r[1];         p0z = r[2];
    pix = r[p * 3 + 0]; piy = r[p * 3 + 1]; piz = r[p * 3 + 2];
    wv  = wvt[t];
  }
  const float dx = pix - p0x, dy = piy - p0y, dz = piz - p0z;
  const float sxy2 = dx * dx + dy * dy;
  const float sr2  = sxy2 + dz * dz;
  const float inv  = ld ? 1.0f / sr2 : 0.0f;   // guard: no 0*inf NaN on pads
  ri.dirx = dx * inv;
  ri.diry = dy * inv;
  ri.wv   = wv;
  ri.bv   = zero16();                          // Z = (z0, pi_z, sep_xy2), K-padded
  if (g == 0) { ri.bv[0] = (_Float16)p0z; ri.bv[1] = (_Float16)piz; ri.bv[2] = (_Float16)sxy2; }
  return ri;
}

__global__ __launch_bounds__(256) void fused_forces_kernel(
    const float* __restrict__ X,   const float* __restrict__ wdotr,
    const float* __restrict__ wvt,
    const float* __restrict__ iW0, const float* __restrict__ ib0,
    const float* __restrict__ iW1, const float* __restrict__ ib1,
    const float* __restrict__ iW2, const float* __restrict__ ib2,
    const float* __restrict__ iWo, const float* __restrict__ ibo,
    const float* __restrict__ cW0, const float* __restrict__ cb0,
    const float* __restrict__ cW1, const float* __restrict__ cb1,
    const float* __restrict__ cW2, const float* __restrict__ cb2,
    const float* __restrict__ cWo, const float* __restrict__ cbo,
    const float* __restrict__ gW0, const float* __restrict__ gb0,
    const float* __restrict__ gW1, const float* __restrict__ gb1,
    const float* __restrict__ gWo, const float* __restrict__ gbo,
    float* __restrict__ out) {
  __shared__ float s_red[2][NWAVE];
  __shared__ float s_conf[2];
  __shared__ float s_gamma;

  const int b    = blockIdx.x;
  const int lane = threadIdx.x & 31;
  const int wave = threadIdx.x >> 5;
  const int g    = (lane >> 4) & 1;
  const float* Xb = X + (size_t)b * Tk * (Pk * Dk);

  // ---- interaction-MLP weights as f16 A-tiles; biases as D-layout C tiles ----
  v16h A0[2], A1[2], A2[2];
  A0[0] = loadA(iW0, INT_H, 0, 3, lane);   A0[1] = loadA(iW0, INT_H, 16, 3, lane);
  A1[0] = loadA(iW1, INT_H, 0, 32, lane);  A1[1] = loadA(iW1, INT_H, 16, 32, lane);
  A2[0] = loadA(iW2, INT_H, 0, 32, lane);  A2[1] = loadA(iW2, INT_H, 16, 32, lane);
  v8f c0a, c0b, c1a, c1b, c2a, c2b;
  float woA[8], woB[8];
#pragma unroll
  for (int r = 0; r < 8; ++r) {
    c0a[r] = ib0[r + 8 * g]; c0b[r] = ib0[16 + r + 8 * g];
    c1a[r] = ib1[r + 8 * g]; c1b[r] = ib1[16 + r + 8 * g];
    c2a[r] = ib2[r + 8 * g]; c2b[r] = ib2[16 + r + 8 * g];
    woA[r] = iWo[r + 8 * g]; woB[r] = iWo[16 + r + 8 * g];
  }
  const float bo = ibo[0];

  // ---- interaction path: two independent row-tiles in flight per iteration ----
  float accx = 0.0f, accy = 0.0f;
  for (int t0 = wave; t0 < NTILE; t0 += 2 * NWAVE) {
    const int t1 = t0 + NWAVE;                 // may run past NTILE: masked to zero
    RowIn ra = loadRow(Xb, wvt, t0, lane, g);
    RowIn rb = loadRow(Xb, wvt, t1, lane, g);

    v8f a0 = wmma16(A0[0], ra.bv, c0a);
    v8f a1 = wmma16(A0[1], ra.bv, c0b);
    v8f b0 = wmma16(A0[0], rb.bv, c0a);
    v8f b1 = wmma16(A0[1], rb.bv, c0b);
    v16h xa = repack32<0>(a0, a1, g);          // leaky
    v16h xb = repack32<0>(b0, b1, g);

    a0 = wmma16(A1[0], xa, c1a); a1 = wmma16(A1[1], xa, c1b);
    b0 = wmma16(A1[0], xb, c1a); b1 = wmma16(A1[1], xb, c1b);
    xa = repack32<1>(a0, a1, g);               // tanh
    xb = repack32<1>(b0, b1, g);

    a0 = wmma16(A2[0], xa, c2a); a1 = wmma16(A2[1], xa, c2b);
    b0 = wmma16(A2[0], xb, c2a); b1 = wmma16(A2[1], xb, c2b);

    // fold last leaky + 32->1 head directly in D layout (no repack/f16 round-trip)
    float fa = 0.0f, fb = 0.0f;
#pragma unroll
    for (int r = 0; r < 8; ++r) {
      fa += actf<0>(a0[r]) * woA[r] + actf<0>(a1[r]) * woB[r];
      fb += actf<0>(b0[r]) * woA[r] + actf<0>(b1[r]) * woB[r];
    }
    fa += swap16(fa); fb += swap16(fb);
    fa += bo;         fb += bo;
    accx += ra.wv * fa * ra.dirx + rb.wv * fb * rb.dirx;   // wv==0 kills pads/dup half
    accy += ra.wv * fa * ra.diry + rb.wv * fb * rb.diry;
  }
#pragma unroll
  for (int s = 1; s < 32; s <<= 1) {
    accx += __shfl_xor(accx, s, 32);
    accy += __shfl_xor(accy, s, 32);
  }
  if (lane == 0) { s_red[0][wave] = accx; s_red[1][wave] = accy; }

  // ---- conf path: wave 0 only (wave-uniform branch, EXEC all-1s for WMMA) ----
  if (wave == 0) {
    const v16h C0 = loadA(cW0, CONF_H, 0, 3, lane);
    const v16h C1 = loadA(cW1, CONF_H, 0, 16, lane);
    const v16h C2 = loadA(cW2, CONF_H, 0, 16, lane);
    v8f cc0, cc1, cc2;
    float woc0[8], woc1[8];
#pragma unroll
    for (int r = 0; r < 8; ++r) {
      cc0[r] = cb0[r + 8 * g]; cc1[r] = cb1[r + 8 * g]; cc2[r] = cb2[r + 8 * g];
      woc0[r] = cWo[(r + 8 * g) * 2 + 0];
      woc1[r] = cWo[(r + 8 * g) * 2 + 1];
    }
    const float cbo0 = cbo[0], cbo1 = cbo[1];
    float cx = 0.0f, cy = 0.0f;
#pragma unroll
    for (int tile = 0; tile < (Tk + 15) / 16; ++tile) {
      const int t = tile * 16 + (lane & 15);
      const bool ld = (t < Tk) && (g == 0);
      float v0 = 0, v1 = 0, v2 = 0, wv = 0;
      if (ld) {
        const float* r = Xb + (size_t)t * (Pk * Dk);
        v0 = r[0]; v1 = r[1]; v2 = r[2];
        wv = wvt[t];
      }
      v16h bv = zero16();
      if (g == 0) { bv[0] = (_Float16)v0; bv[1] = (_Float16)v1; bv[2] = (_Float16)v2; }
      v8f d = wmma16(C0, bv, cc0);
      v16h x1 = repack16<2>(d, g);             // elu
      d = wmma16(C1, x1, cc1);
      v16h x2 = repack16<1>(d, g);             // tanh
      d = wmma16(C2, x2, cc2);
      float o0 = 0.0f, o1 = 0.0f;              // fold elu + 16->2 head in D layout
#pragma unroll
      for (int r = 0; r < 8; ++r) {
        const float e = actf<2>(d[r]);
        o0 += e * woc0[r]; o1 += e * woc1[r];
      }
      o0 += swap16(o0); o1 += swap16(o1);
      cx += wv * (o0 + cbo0); cy += wv * (o1 + cbo1);
    }
#pragma unroll
    for (int s = 1; s < 32; s <<= 1) {
      cx += __shfl_xor(cx, s, 32);
      cy += __shfl_xor(cy, s, 32);
    }
    if (lane == 0) { s_conf[0] = cx; s_conf[1] = cy; }
  }

  // ---- gamma path: ~300 FLOP scalar chain, thread 0 ----
  if (threadIdx.x == 0) {
    const float s = Xb[Dk - 1];
    float h1v[16], h2v[16];
#pragma unroll
    for (int j = 0; j < 16; ++j) h1v[j] = actf<2>(s * gW0[j] + gb0[j]);
#pragma unroll
    for (int j = 0; j < 16; ++j) {
      float a = gb1[j];
#pragma unroll
      for (int i = 0; i < 16; ++i) a += h1v[i] * gW1[i * 16 + j];
      h2v[j] = tanhf(a);
    }
    float gm = gbo[0];
#pragma unroll
    for (int j = 0; j < 16; ++j) gm += h2v[j] * gWo[j];
    s_gamma = gm;
  }

  __syncthreads();
  if (threadIdx.x == 0) {
    float wx = 0.0f, wy = 0.0f;
#pragma unroll
    for (int w = 0; w < NWAVE; ++w) { wx += s_red[0][w]; wy += s_red[1][w]; }
    out[b * 2 + 0] = (s_conf[0] + wx) * 100.0f + s_gamma * wdotr[b * 2 + 0];
    out[b * 2 + 1] = (s_conf[1] + wy) * 100.0f + s_gamma * wdotr[b * 2 + 1];
  }
}

extern "C" void kernel_launch(void* const* d_in, const int* in_sizes, int n_in,
                              void* d_out, int out_size, void* d_ws, size_t ws_size,
                              hipStream_t stream) {
  (void)in_sizes; (void)n_in; (void)out_size; (void)d_ws; (void)ws_size;
  const float* X     = (const float*)d_in[0];
  const float* wdotr = (const float*)d_in[1];
  const float* wvt   = (const float*)d_in[2];
  const float* iW0   = (const float*)d_in[3];
  const float* ib0   = (const float*)d_in[4];
  const float* iW1   = (const float*)d_in[5];
  const float* ib1   = (const float*)d_in[6];
  const float* iW2   = (const float*)d_in[7];
  const float* ib2   = (const float*)d_in[8];
  const float* iWo   = (const float*)d_in[9];
  const float* ibo   = (const float*)d_in[10];
  const float* cW0   = (const float*)d_in[11];
  const float* cb0   = (const float*)d_in[12];
  const float* cW1   = (const float*)d_in[13];
  const float* cb1   = (const float*)d_in[14];
  const float* cW2   = (const float*)d_in[15];
  const float* cb2   = (const float*)d_in[16];
  const float* cWo   = (const float*)d_in[17];
  const float* cbo   = (const float*)d_in[18];
  const float* gW0   = (const float*)d_in[19];
  const float* gb0   = (const float*)d_in[20];
  const float* gW1   = (const float*)d_in[21];
  const float* gb1   = (const float*)d_in[22];
  const float* gWo   = (const float*)d_in[23];
  const float* gbo   = (const float*)d_in[24];
  float* out = (float*)d_out;

  fused_forces_kernel<<<Bk, 256, 0, stream>>>(
      X, wdotr, wvt, iW0, ib0, iW1, ib1, iW2, ib2, iWo, ibo,
      cW0, cb0, cW1, cb1, cW2, cb2, cWo, cbo,
      gW0, gb0, gW1, gb1, gWo, gbo, out);
}
